// ABCGRU_layer_72576357368188
// MI455X (gfx1250) — matrix operand order repaired and checked by reading, not compiled, not run on hardware
//
#include <hip/hip_runtime.h>

#define DEVINL __device__ __forceinline__

constexpr int B  = 8;
constexpr int N  = 2048;
constexpr int C  = 64;
constexpr int Hd = 128;

// LDS row paddings (elements): odd multiples of 8 elems (16B) -> conflict-free b128 reads
constexpr int KPAD = Hd + 8;   // 136 elems = 272B rows for 32x128 key tiles
constexpr int XPAD = 32 + 8;   // 40 elems = 80B rows for 64x32 Xt tiles
constexpr int PPAD = 32 + 8;   // 40 elems for 16x32 P tiles
constexpr int TPAD = Hd + 8;   // 136 elems for 16x128 transpose tiles

typedef __bf16 bf16;
typedef bf16  v16bf __attribute__((ext_vector_type(16)));
typedef bf16  v8bf  __attribute__((ext_vector_type(8)));
typedef float v8f   __attribute__((ext_vector_type(8)));

DEVINL v8f wmma_bf16(v16bf a, v16bf b, v8f c) {
  return __builtin_amdgcn_wmma_f32_16x16x32_bf16(false, a, false, b, (short)0, c, false, false);
}

DEVINL v16bf cat8(v8bf lo, v8bf hi) {
  return __builtin_shufflevector(lo, hi, 0,1,2,3,4,5,6,7,8,9,10,11,12,13,14,15);
}

// A-fragment (16x32 bf16, MxK) from row-major source at (row0,k0), leading dim ld.
// ISA 7.12.2: lane L row=L&15; lanes 0-15 hold K {0..7,16..23}, lanes 16-31 K {8..15,24..31}.
DEVINL v16bf load_aF(const bf16* p, int ld, int lane) {
  int row = lane & 15;
  int h8  = (lane & 16) >> 1;                 // 0 or 8
  const bf16* base = p + (size_t)row * ld + h8;
  v8bf c0 = *(const v8bf*)(base);
  v8bf c1 = *(const v8bf*)(base + 16);
  return cat8(c0, c1);
}

// B-fragment (32x16 bf16, KxN) where B = src^T, i.e. B[k][n] = src[n][k]; src row-major, ld elems.
// Lane L col n=L&15; lanes 0-15 hold K {0..7,16..23}, lanes 16-31 hold K {8..15,24..31}.
DEVINL v16bf load_bT(const bf16* src, int ld, int lane) {
  int n  = lane & 15;
  int kb = (lane & 16) >> 1;                   // 0 or 8
  const bf16* base = src + (size_t)n * ld + kb;
  v8bf c0 = *(const v8bf*)(base);
  v8bf c1 = *(const v8bf*)(base + 16);
  return cat8(c0, c1);
}

DEVINL v8f vzero() { v8f z = {0.f,0.f,0.f,0.f,0.f,0.f,0.f,0.f}; return z; }
DEVINL float sigf(float x) { return 1.0f / (1.0f + __expf(-x)); }

// Generic shared pointers carry the LDS byte address in their low 32 bits (flat aperture rule).
DEVINL unsigned lds_off_u32(const void* p) { return (unsigned)(uintptr_t)p; }

// Async global -> LDS copy of 16B per lane, GVS mode (SGPR base + 32-bit VGPR byte offset).
// Tracked by ASYNCcnt (ISA ch.10 async memory ops).
DEVINL void async_b128(unsigned ldsoff, unsigned voff, const void* sbase) {
  asm volatile("global_load_async_to_lds_b128 %0, %1, %2"
               :: "v"(ldsoff), "v"(voff), "s"(sbase) : "memory");
}

// ---------------------------------------------------------------- K0: fp32 -> bf16
__global__ void k0_convert(const float* __restrict__ X, const float* __restrict__ Hp,
    const float* wxr, const float* wxz, const float* wxh, const float* whh,
    const float* wha1, const float* wha2, const float* wy,
    bf16* Xbf, bf16* Xtbf, bf16* Hbf,
    bf16* Wxr, bf16* Wxz, bf16* Wxh, bf16* Whh, bf16* Wha1, bf16* Wha2, bf16* Wy)
{
  size_t i0 = (size_t)blockIdx.x * blockDim.x + threadIdx.x;
  size_t stride = (size_t)gridDim.x * blockDim.x;
  for (size_t j = i0; j < (size_t)B*N*C; j += stride) {
    float v = X[j];
    Xbf[j] = (bf16)v;
    size_t c = j % C, bn = j / C, n = bn % N, b = bn / N;
    Xtbf[(b*C + c)*N + n] = (bf16)v;           // transposed copy for P@X B-fragments
  }
  for (size_t j = i0; j < (size_t)B*N*Hd; j += stride) Hbf[j] = (bf16)Hp[j];
  for (size_t j = i0; j < (size_t)Hd*C;  j += stride) {
    Wxr[j] = (bf16)wxr[j]; Wxz[j] = (bf16)wxz[j];
    Wxh[j] = (bf16)wxh[j]; Wha1[j] = (bf16)wha1[j];
  }
  for (size_t j = i0; j < (size_t)Hd*Hd; j += stride) {
    Whh[j] = (bf16)whh[j]; Wha2[j] = (bf16)wha2[j];
  }
  for (size_t j = i0; j < (size_t)C*Hd;  j += stride) Wy[j] = (bf16)wy[j];
}

// ---------------------------------------------------------------- K1: lhs/rhs projections
__global__ void __launch_bounds__(128) k1_proj(
    const bf16* __restrict__ Xbf, const bf16* __restrict__ Hbf,
    const bf16* __restrict__ Wha1, const bf16* __restrict__ Wha2,
    const float* __restrict__ bha2,
    bf16* __restrict__ Lbf, bf16* __restrict__ Rbf)
{
  int lane = threadIdx.x & 31;
  int t = blockIdx.x * 4 + (threadIdx.x >> 5);
  int b = t >> 7;                               // N/16 = 128 tiles per batch
  int row0 = (t & 127) << 4;
  int col = lane & 15, rb = (lane >> 4) << 3;

  v16bf xf[2], hf[4];
#pragma unroll
  for (int c = 0; c < 2; ++c) xf[c] = load_aF(Xbf + ((size_t)b*N + row0)*C + c*32, C, lane);
#pragma unroll
  for (int c = 0; c < 4; ++c) hf[c] = load_aF(Hbf + ((size_t)b*N + row0)*Hd + c*32, Hd, lane);

#pragma unroll
  for (int ht = 0; ht < 8; ++ht) {              // lhs = X @ W_ha1^T   (no bias)
    v8f a = vzero();
#pragma unroll
    for (int c = 0; c < 2; ++c)
      a = wmma_bf16(xf[c], load_bT(Wha1 + (size_t)ht*16*C + c*32, C, lane), a);
#pragma unroll
    for (int r = 0; r < 8; ++r)
      Lbf[((size_t)b*N + row0 + r + rb)*Hd + ht*16 + col] = (bf16)a[r];
  }
#pragma unroll
  for (int ht = 0; ht < 8; ++ht) {              // rhs = H_pre @ W_ha2^T + b_ha2
    v8f a = vzero();
#pragma unroll
    for (int c = 0; c < 4; ++c)
      a = wmma_bf16(hf[c], load_bT(Wha2 + (size_t)ht*16*Hd + c*32, Hd, lane), a);
    float bias = bha2[ht*16 + col];
#pragma unroll
    for (int r = 0; r < 8; ++r)
      Rbf[((size_t)b*N + row0 + r + rb)*Hd + ht*16 + col] = (bf16)(a[r] + bias);
  }
}

// ---------------------------------------------------------------- K2: fused flash attention
// 4 waves/block share one batch; key/X tiles staged once per block via async LDS copies,
// double-buffered so the DMA of tile m+32 overlaps the 12 WMMAs of tile m.
__global__ void __launch_bounds__(128) k2_attn(
    const bf16* __restrict__ Lbf, const bf16* __restrict__ Rbf,
    const bf16* __restrict__ Xtbf, bf16* __restrict__ Abf)
{
  __shared__ __align__(16) bf16 sK[2][32*KPAD];  // rhs key tiles  (2 x 8.5 KB)
  __shared__ __align__(16) bf16 sX[2][C*XPAD];   // Xt tiles [ch][key] (2 x 5 KB)
  __shared__ __align__(16) bf16 sP[4][16*PPAD];  // per-wave P transpose buffers

  int tid  = threadIdx.x;
  int lane = tid & 31;
  int w = tid >> 5;
  int t = blockIdx.x * 4 + w;
  int b = t >> 7;
  int row0 = (t & 127) << 4;
  int col = lane & 15, rb = (lane >> 4) << 3;
  bf16* sp = sP[w];

  const bf16* Rbase = Rbf  + (size_t)b*N*Hd;
  const bf16* Xbase = Xtbf + (size_t)b*C*N;

  // Stage one 32-key tile (keys m..m+31): 6 async b128 copies per thread.
  auto stage = [&](int buf, int m) {
    const void* gr = (const void*)(Rbase + (size_t)m*Hd);   // 32 rows x 256B, contiguous
    unsigned lk = lds_off_u32(sK[buf]);
#pragma unroll
    for (int k = 0; k < 4; ++k) {
      int j = tid + k*128;                 // 512 chunks of 16B
      unsigned row = (unsigned)(j >> 4), c16 = (unsigned)(j & 15) * 16u;
      async_b128(lk + row*(KPAD*2) + c16, row*256u + c16, gr);
    }
    const void* gx = (const void*)(Xbase + m);              // 64 rows x 64B, stride N*2
    unsigned lx = lds_off_u32(sX[buf]);
#pragma unroll
    for (int k = 0; k < 2; ++k) {
      int j = tid + k*128;                 // 256 chunks of 16B
      unsigned row = (unsigned)(j >> 2), c16 = (unsigned)(j & 3) * 16u;
      async_b128(lx + row*(XPAD*2) + c16, row*(unsigned)(N*2) + c16, gx);
    }
  };

  v16bf q[4];
#pragma unroll
  for (int c = 0; c < 4; ++c) q[c] = load_aF(Lbf + ((size_t)b*N + row0)*Hd + c*32, Hd, lane);

  v8f acc[4] = {vzero(), vzero(), vzero(), vzero()};   // 16 x 64 output accumulator
  float rmax[8], rsum[8];
#pragma unroll
  for (int r = 0; r < 8; ++r) { rmax[r] = -3.0e38f; rsum[r] = 0.0f; }

  stage(0, 0);
  int p = 0;
  for (int m = 0; m < N; m += 32, p ^= 1) {
    if (m + 32 < N) {
      stage(p ^ 1, m + 32);                               // prefetch next tile
      asm volatile("s_wait_asynccnt 0x6" ::: "memory");   // current tile landed
    } else {
      asm volatile("s_wait_asynccnt 0x0" ::: "memory");
    }
    __syncthreads();                                      // all waves' copies visible

    const bf16* kt0 = sK[p];                              // keys m..m+15
    const bf16* kt1 = sK[p] + 16*KPAD;                    // keys m+16..m+31
    v8f s0 = vzero(), s1 = vzero();
#pragma unroll
    for (int c = 0; c < 4; ++c) s0 = wmma_bf16(q[c], load_bT(kt0 + c*32, KPAD, lane), s0);
#pragma unroll
    for (int c = 0; c < 4; ++c) s1 = wmma_bf16(q[c], load_bT(kt1 + c*32, KPAD, lane), s1);

#pragma unroll
    for (int r = 0; r < 8; ++r) {
      float a0 = s0[r], a1 = s1[r];
      a0 = a0 > 0.f ? a0 : 0.01f * a0;                    // leaky_relu
      a1 = a1 > 0.f ? a1 : 0.01f * a1;
      float mx = fmaxf(a0, a1);
#pragma unroll
      for (int o = 8; o >= 1; o >>= 1) mx = fmaxf(mx, __shfl_xor(mx, o, 32));
      float nm = fmaxf(rmax[r], mx);
      float corr = __expf(rmax[r] - nm);
      rmax[r] = nm;
      float p0 = __expf(a0 - nm), p1 = __expf(a1 - nm);
      float ps = p0 + p1;
#pragma unroll
      for (int o = 8; o >= 1; o >>= 1) ps += __shfl_xor(ps, o, 32);
      rsum[r] = rsum[r] * corr + ps;
#pragma unroll
      for (int u = 0; u < 4; ++u) acc[u][r] *= corr;
      sp[(r + rb)*PPAD + col]      = (bf16)p0;            // P -> LDS (C/D layout scatter)
      sp[(r + rb)*PPAD + 16 + col] = (bf16)p1;
    }
    v16bf pf = load_aF(sp, PPAD, lane);                   // P as A-fragment (in-order DS)
#pragma unroll
    for (int ct = 0; ct < 4; ++ct)
      acc[ct] = wmma_bf16(pf, load_bT(sX[p] + ct*16*XPAD, XPAD, lane), acc[ct]);

    __syncthreads();                                      // protect buffer reuse
  }

#pragma unroll
  for (int r = 0; r < 8; ++r) {
    float inv = 1.0f / rsum[r];
#pragma unroll
    for (int ct = 0; ct < 4; ++ct)
      Abf[((size_t)b*N + row0 + r + rb)*C + ct*16 + col] = (bf16)(acc[ct][r] * inv);
  }
}

// ---------------------------------------------------------------- K3: GRU + output projection
__global__ void __launch_bounds__(128) k3_gru(
    const float* __restrict__ Hpre, const bf16* __restrict__ Abf, const bf16* __restrict__ Xbf,
    const bf16* __restrict__ Wxr, const float* __restrict__ bxr,
    const bf16* __restrict__ Wxz, const float* __restrict__ bxz,
    const bf16* __restrict__ Wxh, const float* __restrict__ bxh,
    const bf16* __restrict__ Whh, const float* __restrict__ bhh,
    const bf16* __restrict__ Wy,  const float* __restrict__ by,
    float* __restrict__ yout, float* __restrict__ hout)
{
  __shared__ __align__(16) bf16 sT[4][16*TPAD];  // per-wave transpose buffers
  int lane = threadIdx.x & 31;
  int w = threadIdx.x >> 5;
  int t = blockIdx.x * 4 + w;
  int b = t >> 7;
  int row0 = (t & 127) << 4;
  int col = lane & 15, rb = (lane >> 4) << 3;
  bf16* st = sT[w];

  v16bf af[2], xf[2];
#pragma unroll
  for (int c = 0; c < 2; ++c) {
    af[c] = load_aF(Abf + ((size_t)b*N + row0)*C + c*32, C, lane);
    xf[c] = load_aF(Xbf + ((size_t)b*N + row0)*C + c*32, C, lane);
  }

  float hp[8][8];                                // H_pre tile in C/D layout
#pragma unroll
  for (int ht = 0; ht < 8; ++ht)
#pragma unroll
    for (int r = 0; r < 8; ++r)
      hp[ht][r] = Hpre[((size_t)b*N + row0 + r + rb)*Hd + ht*16 + col];

  // Rt = sigmoid(A@W_xr^T + b);  RH = Rt * H_pre -> LDS (bf16, row-major)
#pragma unroll
  for (int ht = 0; ht < 8; ++ht) {
    v8f a = vzero();
#pragma unroll
    for (int c = 0; c < 2; ++c)
      a = wmma_bf16(af[c], load_bT(Wxr + (size_t)ht*16*C + c*32, C, lane), a);
    float bias = bxr[ht*16 + col];
#pragma unroll
    for (int r = 0; r < 8; ++r) {
      float rt = sigf(a[r] + bias);
      st[(r + rb)*TPAD + ht*16 + col] = (bf16)(rt * hp[ht][r]);
    }
  }
  v16bf rhf[4];
#pragma unroll
  for (int c = 0; c < 4; ++c) rhf[c] = load_aF(st + c*32, TPAD, lane);

  // H_cand = tanh(X@W_xh^T + RH@W_hh^T + b_xh + b_hh)
  float hc[8][8];
#pragma unroll
  for (int ht = 0; ht < 8; ++ht) {
    v8f a = vzero();
#pragma unroll
    for (int c = 0; c < 2; ++c)
      a = wmma_bf16(xf[c], load_bT(Wxh + (size_t)ht*16*C + c*32, C, lane), a);
#pragma unroll
    for (int c = 0; c < 4; ++c)
      a = wmma_bf16(rhf[c], load_bT(Whh + (size_t)ht*16*Hd + c*32, Hd, lane), a);
    float bias = bxh[ht*16 + col] + bhh[ht*16 + col];
#pragma unroll
    for (int r = 0; r < 8; ++r) hc[ht][r] = tanhf(a[r] + bias);
  }

  // Zt, blend, store H, stage H in LDS for output projection
#pragma unroll
  for (int ht = 0; ht < 8; ++ht) {
    v8f a = vzero();
#pragma unroll
    for (int c = 0; c < 2; ++c)
      a = wmma_bf16(af[c], load_bT(Wxz + (size_t)ht*16*C + c*32, C, lane), a);
    float bias = bxz[ht*16 + col];
#pragma unroll
    for (int r = 0; r < 8; ++r) {
      float z = sigf(a[r] + bias);
      float h = z * hp[ht][r] + (1.0f - z) * hc[ht][r];
      hout[((size_t)b*N + row0 + r + rb)*Hd + ht*16 + col] = h;
      st[(r + rb)*TPAD + ht*16 + col] = (bf16)h;
    }
  }
  v16bf hf[4];
#pragma unroll
  for (int c = 0; c < 4; ++c) hf[c] = load_aF(st + c*32, TPAD, lane);

  // y = H @ W_y^T + b_y
#pragma unroll
  for (int ct = 0; ct < 4; ++ct) {
    v8f a = vzero();
#pragma unroll
    for (int c = 0; c < 4; ++c)
      a = wmma_bf16(hf[c], load_bT(Wy + (size_t)ct*16*Hd + c*32, Hd, lane), a);
    float bias = by[ct*16 + col];
#pragma unroll
    for (int r = 0; r < 8; ++r)
      yout[((size_t)b*N + row0 + r + rb)*C + ct*16 + col] = a[r] + bias;
  }
}

// ---------------------------------------------------------------- host launch
extern "C" void kernel_launch(void* const* d_in, const int* in_sizes, int n_in,
                              void* d_out, int out_size, void* d_ws, size_t ws_size,
                              hipStream_t stream)
{
  (void)in_sizes; (void)n_in; (void)out_size; (void)ws_size;
  const float* Hpre = (const float*)d_in[0];
  const float* X    = (const float*)d_in[1];
  const float* wxr  = (const float*)d_in[2];  const float* bxr  = (const float*)d_in[3];
  const float* wxz  = (const float*)d_in[4];  const float* bxz  = (const float*)d_in[5];
  const float* wxh  = (const float*)d_in[6];  const float* bxh  = (const float*)d_in[7];
  const float* whh  = (const float*)d_in[8];  const float* bhh  = (const float*)d_in[9];
  const float* wha1 = (const float*)d_in[10];
  const float* wha2 = (const float*)d_in[11]; const float* bha2 = (const float*)d_in[12];
  const float* wy   = (const float*)d_in[13]; const float* by   = (const float*)d_in[14];

  char* ws = (char*)d_ws;
  size_t off = 0;
  auto alloc = [&](size_t bytes) -> void* {
    void* p = ws + off;
    off = (off + bytes + 255) & ~(size_t)255;
    return p;
  };
  bf16* Xbf  = (bf16*)alloc((size_t)B*N*C  * 2);
  bf16* Xtbf = (bf16*)alloc((size_t)B*C*N  * 2);
  bf16* Hbf  = (bf16*)alloc((size_t)B*N*Hd * 2);
  bf16* Lbf  = (bf16*)alloc((size_t)B*N*Hd * 2);
  bf16* Rbf  = (bf16*)alloc((size_t)B*N*Hd * 2);
  bf16* Abf  = (bf16*)alloc((size_t)B*N*C  * 2);
  bf16* Wxr  = (bf16*)alloc((size_t)Hd*C   * 2);
  bf16* Wxz  = (bf16*)alloc((size_t)Hd*C   * 2);
  bf16* Wxh  = (bf16*)alloc((size_t)Hd*C   * 2);
  bf16* Whh  = (bf16*)alloc((size_t)Hd*Hd  * 2);
  bf16* Wha1 = (bf16*)alloc((size_t)Hd*C   * 2);
  bf16* Wha2 = (bf16*)alloc((size_t)Hd*Hd  * 2);
  bf16* Wy   = (bf16*)alloc((size_t)C*Hd   * 2);

  k0_convert<<<512, 256, 0, stream>>>(X, Hpre, wxr, wxz, wxh, whh, wha1, wha2, wy,
      Xbf, Xtbf, Hbf, Wxr, Wxz, Wxh, Whh, Wha1, Wha2, Wy);

  int tiles  = B * (N / 16);     // 1024 wave-tiles
  int blocks = tiles / 4;        // 4 waves per 128-thread block

  k1_proj<<<blocks, 128, 0, stream>>>(Xbf, Hbf, Wha1, Wha2, bha2, Lbf, Rbf);
  k2_attn<<<blocks, 128, 0, stream>>>(Lbf, Rbf, Xtbf, Abf);

  float* yout = (float*)d_out;
  float* hout = yout + (size_t)B*N*C;
  k3_gru<<<blocks, 128, 0, stream>>>(Hpre, Abf, Xbf, Wxr, bxr, Wxz, bxz, Wxh, bxh,
                                     Whh, bhh, Wy, by, yout, hout);
}
